// AdaptiveFeaturePropagation_38448547234472
// MI455X (gfx1250) — matrix operand order repaired and checked by reading, not compiled, not run on hardware
//
#include <hip/hip_runtime.h>
#include <hip/hip_bf16.h>

// ---------------------------------------------------------------------------
// AdaptiveFeaturePropagation for MI455X (gfx1250, wave32, WMMA)
//   B=8, C_LOW=128, C_HIGH=256, H=W=128, KS=7
// ~470 GFLOP dominated by three 3x3 convs => implicit GEMM on
// v_wmma_f32_16x16x32_bf16. Activations staged as zero-padded NHWC bf16 so
// the WMMA hot loop has NO border predication. Each wave computes a
// 32px x 32ch register tile (4 accumulators); the K loop is software
// pipelined (double-buffered fragments) so WMMAs of step i overlap the
// loads of step i+1 instead of fencing on loadcnt 0.
// ---------------------------------------------------------------------------

typedef __attribute__((ext_vector_type(16))) __bf16 v16bf;
typedef __attribute__((ext_vector_type(8)))  __bf16 v8bf;
typedef __attribute__((ext_vector_type(8)))  float  v8f;

constexpr int IMG_H = 128;
constexpr int IMG_W = 128;
constexpr int NB    = 8;
constexpr int CLOW  = 128;
constexpr int CHIGH = 256;
constexpr int COUT  = 256;      // the two 3x3 convs emit 256 channels
constexpr int KSZ   = 7;
constexpr int K2    = KSZ * KSZ; // 49

// ---------------------------------------------------------------------------
// Layout transforms (one-time, bandwidth-trivial vs the GEMMs)
// ---------------------------------------------------------------------------

// NCHW f32 -> zero-padded NHWC bf16 interior ([B, H+2, W+2, C])
__global__ __launch_bounds__(256) void nchw2nhwc_pad_bf16(const float* __restrict__ in,
                                                          __bf16* __restrict__ out, int C)
{
  const int h = blockIdx.x, b = blockIdx.y;
  const int WP = IMG_W + 2, HP = IMG_H + 2;
  const long HW = (long)IMG_H * IMG_W;
  for (int idx = threadIdx.x; idx < IMG_W * C; idx += 256) {
    const int c = idx / IMG_W;           // reads coalesced along w
    const int w = idx % IMG_W;
    out[(((long)b * HP + h + 1) * WP + (w + 1)) * C + c] =
        (__bf16)in[((long)b * C + c) * HW + (long)h * IMG_W + w];
  }
}

// Zero the 1-pixel border of a padded [B, H+2, W+2, C] bf16 buffer.
__global__ __launch_bounds__(256) void zero_border(__bf16* __restrict__ buf, int C)
{
  const int b = blockIdx.y;
  const int HP = IMG_H + 2, WP = IMG_W + 2;
  const int nb = 2 * WP + 2 * IMG_H;     // 516 border pixels
  __bf16* img = buf + (long)b * HP * WP * C;
  const long tot = (long)nb * C;
  for (long idx = (long)blockIdx.x * 256 + threadIdx.x; idx < tot;
       idx += (long)gridDim.x * 256) {
    const int c = (int)(idx % C);
    const int p = (int)(idx / C);
    int hh, ww;
    if      (p < WP)            { hh = 0;               ww = p; }
    else if (p < 2 * WP)        { hh = HP - 1;          ww = p - WP; }
    else if (p < 2 * WP + IMG_H){ hh = p - 2 * WP + 1;  ww = 0; }
    else                        { hh = p - 2 * WP - IMG_H + 1; ww = WP - 1; }
    img[((long)hh * WP + ww) * C + c] = (__bf16)0.f;
  }
}

// OIHW f32 -> [9][Cout][Cin] bf16  (ci contiguous: B-fragment rows are linear)
__global__ __launch_bounds__(256) void pack_w_oihw(const float* __restrict__ w,
                                                   __bf16* __restrict__ wp,
                                                   int Cout, int Cin)
{
  const long n = (long)Cout * Cin * 9;
  const long idx = (long)blockIdx.x * 256 + threadIdx.x;
  if (idx >= n) return;
  const int  ci = (int)(idx % Cin);
  const long t  = idx / Cin;
  const int  co = (int)(t % Cout);
  const int  o  = (int)(t / Cout);
  wp[idx] = (__bf16)w[((long)co * Cin + ci) * 9 + o];
}

// w3 [49][256] f32 -> [64][256] bf16 (zero pad), b3 [49] -> [64] f32
__global__ __launch_bounds__(256) void pack_w3(const float* __restrict__ w3,
                                               const float* __restrict__ b3,
                                               __bf16* __restrict__ wp,
                                               float* __restrict__ bp)
{
  const int i = blockIdx.x * 256 + threadIdx.x;
  if (i < 64 * 256) {
    const int o = i >> 8, c = i & 255;
    wp[i] = (o < K2) ? (__bf16)w3[o * 256 + c] : (__bf16)0.f;
  }
  if (i < 64) bp[i] = (i < K2) ? b3[i] : 0.f;
}

// ---------------------------------------------------------------------------
// Fragment loaders.
// A fragment (16-bit A 16x32): lane L holds M=L%16,
//   K = (L<16 ? 0..15 : 16..31) -> two contiguous 16B loads.
// B fragment: lane L holds N=L%16, K=(L<16 ? 0..15 : 16..31) -> one 32B load.
// ---------------------------------------------------------------------------
__device__ __forceinline__ v16bf load_afrag(const __bf16* p)
{
  const v8bf c0 = *(const v8bf*)p;
  const v8bf c1 = *(const v8bf*)(p + 8);
  v16bf a;
  #pragma unroll
  for (int i = 0; i < 8; ++i) { a[i] = c0[i]; a[i + 8] = c1[i]; }
  return a;
}

// ---------------------------------------------------------------------------
// Generic KxK (K=1 or 3) conv as shifted GEMMs on bf16 WMMA, bias+ReLU fused.
// Block: 256 thr = 8 waves = WM x WN wave grid; each wave owns a 32px x 32ch
// register tile => 4 accumulators, 4 WMMAs per K-step on 2 A + 2 B fragments.
// ---------------------------------------------------------------------------
template<int CIN, int K33, int WN, int IPAD, int OPAD, int COUTT, bool OUTF32>
__global__ __launch_bounds__(256)
void conv_relu_wmma(const __bf16* __restrict__ act,   // [B, H+2*IPAD, W+2*IPAD, CIN]
                    const __bf16* __restrict__ wp,    // [K33?9:1][COUTT][CIN]
                    const float*  __restrict__ bias,  // [COUTT]
                    void* __restrict__ outv,          // [B, H+2*OPAD, W+2*OPAD, ctot]
                    int ctot, int cbase)
{
  constexpr int WM  = 8 / WN;
  constexpr int BPX = WM * 32;           // block pixel tile (along W)
  constexpr int BCH = WN * 32;           // block channel tile
  constexpr int NT  = COUTT / BCH;       // channel tiles in grid.x
  constexpr int IWP = IMG_W + 2 * IPAD, IHP = IMG_H + 2 * IPAD;
  constexpr int OWP = IMG_W + 2 * OPAD, OHP = IMG_H + 2 * OPAD;

  const int tid  = threadIdx.x;
  const int lane = tid & 31;
  const int wave = tid >> 5;
  const int wn = wave % WN, wm = wave / WN;
  const int nt = blockIdx.x % NT;
  const int wt = blockIdx.x / NT;
  const int h  = blockIdx.y, b = blockIdx.z;

  const int  w0 = wt * BPX + wm * 32;    // this wave's 32-pixel window
  const int  n0 = nt * BCH + wn * 32;    // this wave's 32-channel window
  const int  lm = lane & 15;
  const bool hi = lane >= 16;
  const int  ka  = hi ? 16 : 0;          // A K-half offset
  const int  kbh = hi ? 16 : 0;          // B K-half offset

  const __bf16* img = act + ((long)b * IHP) * IWP * CIN;
  v8f acc[2][2] = {};

  constexpr int RLO = K33 ? 0 : 1, RHI = K33 ? 2 : 1;
  #pragma unroll
  for (int r = RLO; r <= RHI; ++r) {
    const __bf16* row = img + (long)(h + r - 1 + IPAD) * IWP * CIN;
    #pragma unroll
    for (int s = RLO; s <= RHI; ++s) {
      const int o = K33 ? (r * 3 + s) : 0;
      const __bf16* w0p = wp + ((long)o * COUTT + n0 + lm)      * CIN + kbh;
      const __bf16* w1p = wp + ((long)o * COUTT + n0 + 16 + lm) * CIN + kbh;
      const __bf16* a0p = row + (long)(w0 + lm + s - 1 + IPAD) * CIN + ka;
      const __bf16* a1p = a0p + 16 * CIN;

      // Software-pipelined K loop: WMMAs of step i overlap loads of step i+1.
      v16bf b0 = *(const v16bf*)(w0p);
      v16bf b1 = *(const v16bf*)(w1p);
      v16bf a0 = load_afrag(a0p);
      v16bf a1 = load_afrag(a1p);
      #pragma unroll 2
      for (int kb = 32; kb < CIN; kb += 32) {
        const v16bf b0n = *(const v16bf*)(w0p + kb);
        const v16bf b1n = *(const v16bf*)(w1p + kb);
        const v16bf a0n = load_afrag(a0p + kb);
        const v16bf a1n = load_afrag(a1p + kb);
        acc[0][0] = __builtin_amdgcn_wmma_f32_16x16x32_bf16(
            false, a0, false, b0, (short)0, acc[0][0], false, false);
        acc[0][1] = __builtin_amdgcn_wmma_f32_16x16x32_bf16(
            false, a0, false, b1, (short)0, acc[0][1], false, false);
        acc[1][0] = __builtin_amdgcn_wmma_f32_16x16x32_bf16(
            false, a1, false, b0, (short)0, acc[1][0], false, false);
        acc[1][1] = __builtin_amdgcn_wmma_f32_16x16x32_bf16(
            false, a1, false, b1, (short)0, acc[1][1], false, false);
        b0 = b0n; b1 = b1n; a0 = a0n; a1 = a1n;
      }
      acc[0][0] = __builtin_amdgcn_wmma_f32_16x16x32_bf16(
          false, a0, false, b0, (short)0, acc[0][0], false, false);
      acc[0][1] = __builtin_amdgcn_wmma_f32_16x16x32_bf16(
          false, a0, false, b1, (short)0, acc[0][1], false, false);
      acc[1][0] = __builtin_amdgcn_wmma_f32_16x16x32_bf16(
          false, a1, false, b0, (short)0, acc[1][0], false, false);
      acc[1][1] = __builtin_amdgcn_wmma_f32_16x16x32_bf16(
          false, a1, false, b1, (short)0, acc[1][1], false, false);
    }
  }

  // C/D layout: VGPR i -> M = i (lanes 0-15) or 8+i (lanes 16-31); N = lane%16
  const float bv[2] = { bias[n0 + lm], bias[n0 + 16 + lm] };
  #pragma unroll
  for (int t = 0; t < 2; ++t) {
    const int pmb = w0 + t * 16 + (hi ? 8 : 0);
    #pragma unroll
    for (int u = 0; u < 2; ++u) {
      #pragma unroll
      for (int i = 0; i < 8; ++i) {
        float v = acc[t][u][i] + bv[u];
        v = v > 0.f ? v : 0.f;
        const long idx = (((long)b * OHP + h + OPAD) * OWP + (pmb + i + OPAD)) * ctot
                         + cbase + n0 + u * 16 + lm;
        if (OUTF32) ((float*)outv)[idx]  = v;
        else        ((__bf16*)outv)[idx] = (__bf16)v;
      }
    }
  }
}

// ---------------------------------------------------------------------------
// Softmax over the first 49 of 64 f32 scores per pixel -> [B,H,W,49] kernels.
// ---------------------------------------------------------------------------
__global__ __launch_bounds__(256)
void softmax49(const float* __restrict__ s, float* __restrict__ kern)
{
  const long pix = (long)blockIdx.x * 256 + threadIdx.x;   // < B*H*W
  const float* sp = s + pix * 64;
  float mx = -1e30f;
  #pragma unroll
  for (int o = 0; o < K2; ++o) mx = fmaxf(mx, sp[o]);
  float e[K2], sum = 0.f;
  #pragma unroll
  for (int o = 0; o < K2; ++o) { e[o] = __expf(sp[o] - mx); sum += e[o]; }
  const float inv = 1.f / sum;
  float* kp = kern + pix * K2;
  #pragma unroll
  for (int o = 0; o < K2; ++o) kp[o] = e[o] * inv;
}

// ---------------------------------------------------------------------------
// Spatially-variant 7x7 conv over key_high (f32 NCHW), f32 NCHW output.
// 16x16 pixel tile. Each thread's 49 softmax taps are constant across all
// 256 channels => keep them in 49 registers (no LDS reads in the MAC loop);
// only the shared 4-channel halo lives in LDS.
// ---------------------------------------------------------------------------
__global__ __launch_bounds__(256)
void svc7x7(const float* __restrict__ kh,   // [B,256,H,W]
            const float* __restrict__ kern, // [B,H,W,49]
            float* __restrict__ out)        // [B,256,H,W]
{
  __shared__ float halo[4 * 22 * 22];   // 7.6 KB
  const int tid = threadIdx.x;
  const int b = blockIdx.z, h0 = blockIdx.y * 16, w0 = blockIdx.x * 16;
  const int py = tid >> 4, px = tid & 15;

  float kreg[K2];
  {
    const float* kp = kern + (((long)b * IMG_H + h0 + py) * IMG_W + w0 + px) * K2;
    #pragma unroll
    for (int o = 0; o < K2; ++o) kreg[o] = kp[o];
  }

  const float* inb = kh + (long)b * CHIGH * IMG_H * IMG_W;
  for (int c0 = 0; c0 < CHIGH; c0 += 4) {
    __syncthreads();  // previous iteration's halo reads complete
    for (int idx = tid; idx < 4 * 22 * 22; idx += 256) {
      const int cc = idx / 484, rr = idx % 484;
      const int hh = rr / 22,  ww = rr % 22;
      const int gh = h0 + hh - 3, gw = w0 + ww - 3;
      float v = 0.f;
      if ((unsigned)gh < (unsigned)IMG_H && (unsigned)gw < (unsigned)IMG_W)
        v = inb[((long)(c0 + cc) * IMG_H + gh) * IMG_W + gw];
      halo[idx] = v;
    }
    __syncthreads();
    #pragma unroll
    for (int cc = 0; cc < 4; ++cc) {
      const float* hp = halo + cc * 484 + py * 22 + px;
      float a = 0.f;
      #pragma unroll
      for (int i = 0; i < KSZ; ++i)
        #pragma unroll
        for (int j = 0; j < KSZ; ++j)
          a += hp[i * 22 + j] * kreg[i * KSZ + j];
      out[(((long)b * CHIGH + c0 + cc) * IMG_H + h0 + py) * IMG_W + w0 + px] = a;
    }
  }
}

// ---------------------------------------------------------------------------
// Host-side orchestration
// ---------------------------------------------------------------------------
extern "C" void kernel_launch(void* const* d_in, const int* in_sizes, int n_in,
                              void* d_out, int out_size, void* d_ws, size_t ws_size,
                              hipStream_t stream)
{
  const float* cur_low  = (const float*)d_in[0];
  const float* key_low  = (const float*)d_in[1];
  const float* key_high = (const float*)d_in[2];
  const float* w_red    = (const float*)d_in[3];
  const float* b_red    = (const float*)d_in[4];
  const float* w2       = (const float*)d_in[5];
  const float* b2       = (const float*)d_in[6];
  const float* w3       = (const float*)d_in[7];
  const float* b3       = (const float*)d_in[8];
  float* out = (float*)d_out;

  char* ws = (char*)d_ws;
  size_t off = 0;
  auto alloc = [&](size_t bytes) -> void* {
    void* p = ws + off;
    off += (bytes + 255) & ~(size_t)255;
    return p;
  };
  const long PIX  = (long)NB * IMG_H * IMG_W;                        // 131072
  const long PPIX = (long)NB * (IMG_H + 2) * (IMG_W + 2);            // padded
  __bf16* a_cur  = (__bf16*)alloc(PPIX * CLOW * 2);                  // 34.6 MB
  __bf16* a_key  = (__bf16*)alloc(PPIX * CLOW * 2);                  // 34.6 MB
  __bf16* concat = (__bf16*)alloc(PPIX * 512  * 2);                  // 138 MB
  __bf16* x2     = (__bf16*)alloc(PIX  * COUT * 2);                  // 67 MB
  float*  sbuf   = (float*) alloc(PIX  * 64   * 4);                  // 33.5 MB
  float*  kbuf   = (float*) alloc(PIX  * K2   * 4);                  // 25.7 MB
  __bf16* wredp  = (__bf16*)alloc((size_t)9 * COUT * CLOW * 2);
  __bf16* w2p    = (__bf16*)alloc((size_t)9 * COUT * 512  * 2);
  __bf16* w3pad  = (__bf16*)alloc((size_t)64 * 256 * 2);
  float*  b3pad  = (float*) alloc(64 * 4);

  // layout transforms + border zeroing
  nchw2nhwc_pad_bf16<<<dim3(IMG_H, NB), 256, 0, stream>>>(cur_low, a_cur, CLOW);
  nchw2nhwc_pad_bf16<<<dim3(IMG_H, NB), 256, 0, stream>>>(key_low, a_key, CLOW);
  zero_border<<<dim3((516 * CLOW + 255) / 256, NB), 256, 0, stream>>>(a_cur, CLOW);
  zero_border<<<dim3((516 * CLOW + 255) / 256, NB), 256, 0, stream>>>(a_key, CLOW);
  zero_border<<<dim3((516 * 512 + 255) / 256, NB), 256, 0, stream>>>(concat, 512);
  {
    const long n1 = (long)COUT * CLOW * 9;
    pack_w_oihw<<<dim3((unsigned)((n1 + 255) / 256)), 256, 0, stream>>>(w_red, wredp, COUT, CLOW);
    const long n2 = (long)COUT * 512 * 9;
    pack_w_oihw<<<dim3((unsigned)((n2 + 255) / 256)), 256, 0, stream>>>(w2, w2p, COUT, 512);
    pack_w3<<<dim3((64 * 256 + 255) / 256), 256, 0, stream>>>(w3, b3, w3pad, b3pad);
  }

  // conv_red on cur and key -> padded concat (channels [0,256) / [256,512))
  // grid.x = (W/64 pixel tiles) * (256/128 channel tiles) = 4
  const dim3 cg(4, IMG_H, NB);
  conv_relu_wmma<CLOW, 1, 4, 1, 1, COUT, false>
      <<<cg, 256, 0, stream>>>(a_cur, wredp, b_red, concat, 512, 0);
  conv_relu_wmma<CLOW, 1, 4, 1, 1, COUT, false>
      <<<cg, 256, 0, stream>>>(a_key, wredp, b_red, concat, 512, 256);
  // conv2 on padded concat (Cin=512) -> unpadded x2
  conv_relu_wmma<512, 1, 4, 1, 0, COUT, false>
      <<<cg, 256, 0, stream>>>(concat, w2p, b2, x2, 256, 0);
  // 1x1 conv (256 -> 64, upper 15 outputs are zero weights) on WMMA -> sbuf f32
  conv_relu_wmma<256, 0, 2, 0, 0, 64, true>
      <<<dim3(1, IMG_H, NB), 256, 0, stream>>>(x2, w3pad, b3pad, sbuf, 64, 0);

  // softmax over first 49 scores => per-pixel kernels
  softmax49<<<dim3((unsigned)(PIX / 256)), 256, 0, stream>>>(sbuf, kbuf);

  // spatially-variant 7x7 conv
  svc7x7<<<dim3(IMG_W / 16, IMG_H / 16, NB), 256, 0, stream>>>(key_high, kbuf, out);
}
